// AriaTextMoELayer_64544768524491
// MI455X (gfx1250) — compile-verified, matrix operand order checked by reading, not run on hardware
//
#include <hip/hip_runtime.h>
#include <hip/hip_bf16.h>
#include <cstdint>

#define HDIM   1024
#define NEXP   8
#define IDIM   1024
#define ISHDIM 2048
#define NTOPK  2
#define CHUNK  512   // intermediate-activation chunk (columns) kept in LDS

typedef _Float16 v8h  __attribute__((ext_vector_type(8)));
typedef _Float16 v16h __attribute__((ext_vector_type(16)));
typedef float    v8f  __attribute__((ext_vector_type(8)));

// ---- WMMA fragment helpers -------------------------------------------------
// CDNA5 16-bit A 16x32 layout: lanes 0-15 hold M=lane, K={0..7,16..23};
// lanes 16-31 hold M=lane-16, K={8..15,24..31}.  Row data lives in LDS with
// K contiguous, so each lane needs two 16-byte chunks -> 2x ds_load_b128.
__device__ __forceinline__ v16h load_a_frag(const _Float16* row, int kb, int koff) {
  const v8h lo = *(const v8h*)(row + kb + koff);
  const v8h hi = *(const v8h*)(row + kb + 16 + koff);
  v16h r;
#pragma unroll
  for (int i = 0; i < 8; ++i) { r[i] = lo[i]; r[i + 8] = hi[i]; }
  return r;
}

// B 32x16 layout (mirror of A): lane's column n = n0 + (lane&15), K split the
// same way.  Load fp32 weights directly (rows are coalesced across lanes 0-15
// and 16-31 respectively) and convert to f16 in registers.
__device__ __forceinline__ v16h load_b_frag(const float* __restrict__ B, int ldb,
                                            int kbase, int n0, int lane) {
  const int n  = n0 + (lane & 15);
  const int k0 = kbase + ((lane >> 4) << 3);
  const float* p = B + (size_t)k0 * ldb + n;
  v16h r;
#pragma unroll
  for (int i = 0; i < 8; ++i) {
    r[i]     = (_Float16)p[(size_t)i * ldb];
    r[i + 8] = (_Float16)p[(size_t)(i + 16) * ldb];
  }
  return r;
}

__device__ __forceinline__ v8f wmma16(v16h a, v16h b, v8f c) {
  return __builtin_amdgcn_wmma_f32_16x16x32_f16(false, a, false, b, (short)0, c,
                                                false, false);
}

// fast SwiGLU activation: silu(p)*g with v_rcp_f32 instead of IEEE division
__device__ __forceinline__ float swiglu_f(float p, float g) {
  float sig = __builtin_amdgcn_rcpf(1.0f + __expf(-p));
  return p * sig * g;
}

// ---- 1. router: logits, top-2, softmax-of-2, expert counts -----------------
__global__ void router_kernel(const float* __restrict__ x, const float* __restrict__ wr,
                              int* __restrict__ eix, float* __restrict__ esc,
                              int* __restrict__ counts, int ntok) {
  int t = blockIdx.x * blockDim.x + threadIdx.x;
  if (t >= ntok) return;
  const float* row = x + (size_t)t * HDIM;
  float lg[NEXP];
#pragma unroll
  for (int e = 0; e < NEXP; ++e) lg[e] = 0.0f;
  for (int k = 0; k < HDIM; ++k) {
    float xv = row[k];
    const float* w = wr + (size_t)k * NEXP;
#pragma unroll
    for (int e = 0; e < NEXP; ++e) lg[e] = fmaf(xv, w[e], lg[e]);
  }
  int i0 = 0;
#pragma unroll
  for (int e = 1; e < NEXP; ++e) if (lg[e] > lg[i0]) i0 = e;
  int i1 = -1;
#pragma unroll
  for (int e = 0; e < NEXP; ++e)
    if (e != i0 && (i1 < 0 || lg[e] > lg[i1])) i1 = e;
  float ex = __expf(lg[i1] - lg[i0]);   // <= 1
  float p0 = 1.0f / (1.0f + ex);        // router probabilities: keep exact
  float p1 = 1.0f - p0;
  eix[2 * t]     = i0;  eix[2 * t + 1] = i1;
  esc[2 * t]     = p0;  esc[2 * t + 1] = p1;
  atomicAdd(&counts[i0], 1);
  atomicAdd(&counts[i1], 1);
}

// ---- 2. exclusive scan of 8 expert counts ----------------------------------
__global__ void scan_kernel(const int* __restrict__ counts, int* __restrict__ offsets) {
  if (threadIdx.x == 0) {
    int acc = 0;
    for (int e = 0; e < NEXP; ++e) { offsets[e] = acc; acc += counts[e]; }
  }
}

// ---- 3. scatter token/score into contiguous per-expert slots ---------------
__global__ void scatter_kernel(const int* __restrict__ eix, const float* __restrict__ esc,
                               const int* __restrict__ offsets, int* __restrict__ cursors,
                               int* __restrict__ slots, float* __restrict__ slotSc,
                               int ntok) {
  int t = blockIdx.x * blockDim.x + threadIdx.x;
  if (t >= ntok) return;
#pragma unroll
  for (int j = 0; j < NTOPK; ++j) {
    int e   = eix[2 * t + j];
    int pos = offsets[e] + atomicAdd(&cursors[e], 1);
    slots[pos]  = t;
    slotSc[pos] = esc[2 * t + j];
  }
}

// ---- 4. grouped expert SwiGLU MLP (fc1 -> silu*gate -> fc2) ----------------
__global__ void __launch_bounds__(256)
moe_kernel(const float* __restrict__ x, const float* __restrict__ fc1,
           const float* __restrict__ fc2, const int* __restrict__ slots,
           const float* __restrict__ slotSc, const int* __restrict__ counts,
           const int* __restrict__ offsets, float* __restrict__ out) {
  const int e   = blockIdx.y;
  const int cnt = counts[e];
  const int m0  = blockIdx.x * 16;
  if (m0 >= cnt) return;                       // uniform per block

  __shared__ __align__(16) _Float16 Xs[16 * HDIM];   // 32 KB gathered X tile
  __shared__ __align__(16) _Float16 As[16 * CHUNK];  // 16 KB activation chunk
  __shared__ int   tokS[16];
  __shared__ float scS[16];

  const int tid = threadIdx.x;
  if (tid < 16) {
    int i = m0 + tid;
    if (i < cnt) { tokS[tid] = slots[offsets[e] + i]; scS[tid] = slotSc[offsets[e] + i]; }
    else         { tokS[tid] = slots[offsets[e]];     scS[tid] = 0.0f; }  // pad row, score 0
  }
  __syncthreads();

  // stage gathered token rows -> f16 LDS
  for (int m = 0; m < 16; ++m) {
    const float* row = x + (size_t)tokS[m] * HDIM;
    for (int k = tid; k < HDIM; k += 256) Xs[m * HDIM + k] = (_Float16)row[k];
  }

  const int wave = tid >> 5;
  const int lane = tid & 31;
  const int mrow = lane & 15;
  const int hi   = lane >> 4;
  const int koff = hi << 3;

  const float* B1 = fc1 + (size_t)e * HDIM * (2 * IDIM);
  const float* B2 = fc2 + (size_t)e * IDIM * HDIM;

  v8f vzero = {};
  v8f accOut[8];
  for (int i = 0; i < 8; ++i) accOut[i] = vzero;

  for (int chunk = 0; chunk < IDIM / CHUNK; ++chunk) {
    const int cbase = chunk * CHUNK;
    __syncthreads();   // As free to overwrite (also covers X-stage on chunk 0)
    // fc1 + SwiGLU for intermediate columns [cbase, cbase+CHUNK)
    for (int ct = wave; ct < CHUNK / 16; ct += 8) {
      const int c = cbase + ct * 16;
      v8f accP = vzero, accG = vzero;
      for (int kt = 0; kt < HDIM / 32; ++kt) {
        const int kb = kt * 32;
        v16h a  = load_a_frag(Xs + mrow * HDIM, kb, koff);
        v16h bp = load_b_frag(B1, 2 * IDIM, kb, c, lane);         // proj half
        v16h bg = load_b_frag(B1, 2 * IDIM, kb, c + IDIM, lane);  // gate half
        accP = wmma16(a, bp, accP);
        accG = wmma16(a, bg, accG);
      }
      const int n = ct * 16 + (lane & 15);
#pragma unroll
      for (int j = 0; j < 8; ++j)
        As[(j + (hi << 3)) * CHUNK + n] = (_Float16)swiglu_f(accP[j], accG[j]);
    }
    __syncthreads();
    // fc2 partial accumulation over K-range [cbase, cbase+CHUNK)
    for (int i = 0; i < 8; ++i) {
      const int n0 = (wave + i * 8) * 16;
      v8f acc = accOut[i];
      for (int kt = 0; kt < CHUNK / 32; ++kt) {
        const int kb = kt * 32;
        v16h a = load_a_frag(As + mrow * CHUNK, kb, koff);
        v16h b = load_b_frag(B2, HDIM, cbase + kb, n0, lane);
        acc = wmma16(a, b, acc);
      }
      accOut[i] = acc;
    }
  }

  // epilogue: score-scaled scatter-add into output
  for (int i = 0; i < 8; ++i) {
    const int n = (wave + i * 8) * 16 + (lane & 15);
#pragma unroll
    for (int j = 0; j < 8; ++j) {
      const int mm = j + (hi << 3);
      atomicAdd(&out[(size_t)tokS[mm] * HDIM + n], scS[mm] * accOut[i][j]);
    }
  }
}

// ---- 5. shared-expert SwiGLU MLP (all tokens) ------------------------------
__global__ void __launch_bounds__(256)
shared_kernel(const float* __restrict__ x, const float* __restrict__ gw,
              const float* __restrict__ uw, const float* __restrict__ dw,
              float* __restrict__ out) {
  const int t0 = blockIdx.x * 16;

  __shared__ __align__(16) _Float16 Xs[16 * HDIM];
  __shared__ __align__(16) _Float16 As[16 * CHUNK];

  const int tid = threadIdx.x;
  for (int m = 0; m < 16; ++m) {
    const float* row = x + (size_t)(t0 + m) * HDIM;
    for (int k = tid; k < HDIM; k += 256) Xs[m * HDIM + k] = (_Float16)row[k];
  }

  const int wave = tid >> 5;
  const int lane = tid & 31;
  const int mrow = lane & 15;
  const int hi   = lane >> 4;
  const int koff = hi << 3;

  v8f vzero = {};
  v8f accOut[8];
  for (int i = 0; i < 8; ++i) accOut[i] = vzero;

  for (int chunk = 0; chunk < ISHDIM / CHUNK; ++chunk) {
    const int cbase = chunk * CHUNK;
    __syncthreads();
    // silu(x@gate) * (x@up) for columns [cbase, cbase+CHUNK)
    for (int ct = wave; ct < CHUNK / 16; ct += 8) {
      const int c = cbase + ct * 16;
      v8f accG = vzero, accU = vzero;
      for (int kt = 0; kt < HDIM / 32; ++kt) {
        const int kb = kt * 32;
        v16h a  = load_a_frag(Xs + mrow * HDIM, kb, koff);
        v16h bg = load_b_frag(gw, ISHDIM, kb, c, lane);
        v16h bu = load_b_frag(uw, ISHDIM, kb, c, lane);
        accG = wmma16(a, bg, accG);
        accU = wmma16(a, bu, accU);
      }
      const int n = ct * 16 + (lane & 15);
#pragma unroll
      for (int j = 0; j < 8; ++j)
        As[(j + (hi << 3)) * CHUNK + n] = (_Float16)swiglu_f(accG[j], accU[j]);
    }
    __syncthreads();
    // down-projection partial accumulation over K-range [cbase, cbase+CHUNK)
    for (int i = 0; i < 8; ++i) {
      const int n0 = (wave + i * 8) * 16;
      v8f acc = accOut[i];
      for (int kt = 0; kt < CHUNK / 32; ++kt) {
        const int kb = kt * 32;
        v16h a = load_a_frag(As + mrow * CHUNK, kb, koff);
        v16h b = load_b_frag(dw, HDIM, cbase + kb, n0, lane);
        acc = wmma16(a, b, acc);
      }
      accOut[i] = acc;
    }
  }

  for (int i = 0; i < 8; ++i) {
    const int n = (wave + i * 8) * 16 + (lane & 15);
#pragma unroll
    for (int j = 0; j < 8; ++j) {
      const int mm = j + (hi << 3);
      atomicAdd(&out[(size_t)(t0 + mm) * HDIM + n], accOut[i][j]);
    }
  }
}

// ---- launch ----------------------------------------------------------------
extern "C" void kernel_launch(void* const* d_in, const int* in_sizes, int n_in,
                              void* d_out, int out_size, void* d_ws, size_t ws_size,
                              hipStream_t stream) {
  (void)n_in; (void)ws_size;
  const float* x   = (const float*)d_in[0];  // [N, H]
  const float* wr  = (const float*)d_in[1];  // [H, E]
  const float* fc1 = (const float*)d_in[2];  // [E, H, 2I]
  const float* fc2 = (const float*)d_in[3];  // [E, I, H]
  const float* gw  = (const float*)d_in[4];  // [H, ISH]
  const float* uw  = (const float*)d_in[5];  // [H, ISH]
  const float* dw  = (const float*)d_in[6];  // [ISH, H]
  float* out = (float*)d_out;

  const int ntok = in_sizes[0] / HDIM;       // B*S = 2048

  // workspace layout (tiny: ~64 KB)
  char* p = (char*)d_ws;
  int*   counts  = (int*)p;   p += 8 * sizeof(int);
  int*   cursors = (int*)p;   p += 8 * sizeof(int);
  int*   offsets = (int*)p;   p += 8 * sizeof(int);
  int*   eix     = (int*)p;   p += (size_t)2 * ntok * sizeof(int);
  float* esc     = (float*)p; p += (size_t)2 * ntok * sizeof(float);
  int*   slots   = (int*)p;   p += (size_t)2 * ntok * sizeof(int);
  float* slotSc  = (float*)p; p += (size_t)2 * ntok * sizeof(float);

  hipMemsetAsync(out, 0, (size_t)out_size * sizeof(float), stream);
  hipMemsetAsync(counts, 0, 16 * sizeof(int), stream);   // counts + cursors

  router_kernel<<<(ntok + 255) / 256, 256, 0, stream>>>(x, wr, eix, esc, counts, ntok);
  scan_kernel<<<1, 32, 0, stream>>>(counts, offsets);
  scatter_kernel<<<(ntok + 255) / 256, 256, 0, stream>>>(eix, esc, offsets, cursors,
                                                         slots, slotSc, ntok);

  dim3 mgrid((2 * ntok + 15) / 16, NEXP);  // worst case: all slots on one expert
  moe_kernel<<<mgrid, 256, 0, stream>>>(x, fc1, fc2, slots, slotSc, counts, offsets, out);
  shared_kernel<<<(ntok + 15) / 16, 256, 0, stream>>>(x, gw, uw, dw, out);
}
